// GNN_8229157339258
// MI455X (gfx1250) — matrix operand order repaired and checked by reading, not compiled, not run on hardware
//
#include <hip/hip_runtime.h>
#include <hip/hip_bf16.h>
#include <math.h>

typedef __bf16 bf16;
typedef __attribute__((ext_vector_type(16))) __bf16 v16bf;
typedef __attribute__((ext_vector_type(8)))  float  v8f;
typedef int v4i_ __attribute__((vector_size(16)));
typedef __attribute__((address_space(1))) v4i_ g_v4i;   // global int4
typedef __attribute__((address_space(3))) v4i_ l_v4i;   // LDS int4

#define NN 50000
#define NE 800000
#define NF 64

__device__ __forceinline__ float silu_f(float x) { return x / (1.0f + __expf(-x)); }

// ---------------------------------------------------------------------------
// CDNA5 async global->LDS copy (GLOBAL_LOAD_ASYNC_TO_LDS_B128, ASYNCcnt)
// ---------------------------------------------------------------------------
__device__ __forceinline__ void async_copy16(const void* gp, void* lp) {
#if __has_builtin(__builtin_amdgcn_global_load_async_to_lds_b128)
  __builtin_amdgcn_global_load_async_to_lds_b128((g_v4i*)gp, (l_v4i*)lp, 0, 0);
#else
  unsigned lo = (unsigned)(unsigned long)(l_v4i*)lp;
  asm volatile("global_load_async_to_lds_b128 %0, %1, off"
               :: "v"(lo), "v"(gp) : "memory");
#endif
}

template <int N>
__device__ __forceinline__ void wait_async() {
#if __has_builtin(__builtin_amdgcn_s_wait_asynccnt)
  __builtin_amdgcn_s_wait_asynccnt(N);
#else
  asm volatile("s_wait_asynccnt %0" :: "i"(N) : "memory");
#endif
}

// ---------------------------------------------------------------------------
// Column min/max over x[NN, NF] : one block per feature column
// ---------------------------------------------------------------------------
__global__ void colminmax_kernel(const float* __restrict__ x,
                                 float* __restrict__ xmin, float* __restrict__ xmax) {
  int f = blockIdx.x;
  float mn = 3.4e38f, mx = -3.4e38f;
  for (int r = threadIdx.x; r < NN; r += blockDim.x) {
    float v = x[(size_t)r * NF + f];
    mn = fminf(mn, v); mx = fmaxf(mx, v);
  }
  __shared__ float smn[256], smx[256];
  smn[threadIdx.x] = mn; smx[threadIdx.x] = mx;
  __syncthreads();
  for (int s = 128; s > 0; s >>= 1) {
    if (threadIdx.x < s) {
      smn[threadIdx.x] = fminf(smn[threadIdx.x], smn[threadIdx.x + s]);
      smx[threadIdx.x] = fmaxf(smx[threadIdx.x], smx[threadIdx.x + s]);
    }
    __syncthreads();
  }
  if (threadIdx.x == 0) { xmin[f] = smn[0]; xmax[f] = smx[0]; }
}

__global__ void scale_x_kernel(const float* __restrict__ x, const float* __restrict__ xmin,
                               const float* __restrict__ xmax, bf16* __restrict__ out) {
  long gid = (long)blockIdx.x * blockDim.x + threadIdx.x;
  if (gid >= (long)NN * NF) return;
  int f = (int)(gid & (NF - 1));
  out[gid] = (bf16)((x[gid] - xmin[f]) / (xmax[f] - xmin[f]));
}

// ---------------------------------------------------------------------------
// Degree / norm precompute
// ---------------------------------------------------------------------------
__global__ void zero_f32_kernel(float* __restrict__ p, long n) {
  long gid = (long)blockIdx.x * blockDim.x + threadIdx.x;
  if (gid < n) p[gid] = 0.0f;
}

__global__ void deg_kernel(const int* __restrict__ dst, float* __restrict__ deg) {
  long gid = (long)blockIdx.x * blockDim.x + threadIdx.x;
  if (gid < NE) atomicAdd(&deg[dst[gid]], 1.0f);
}

__global__ void dis_kernel(const float* __restrict__ deg, float* __restrict__ dis) {
  long gid = (long)blockIdx.x * blockDim.x + threadIdx.x;
  if (gid < NN) dis[gid] = rsqrtf(fmaxf(deg[gid] + 1.0f, 1e-12f)); // +1 self loop
}

__global__ void norm_kernel(const int* __restrict__ src, const int* __restrict__ dst,
                            const float* __restrict__ dis, float* __restrict__ nrm) {
  long gid = (long)blockIdx.x * blockDim.x + threadIdx.x;
  if (gid < NE) nrm[gid] = dis[src[gid]] * dis[dst[gid]];
}

// ---------------------------------------------------------------------------
// Time embedding: t = relu(emb @ Wt + bt), emb = [sin(ts*f), cos(ts*f)]
// ---------------------------------------------------------------------------
__global__ void time_kernel(const float* __restrict__ ts, const float* __restrict__ W,
                            const float* __restrict__ b, float* __restrict__ tout) {
  __shared__ float emb[128];
  int j = threadIdx.x;
  if (j < 64) {
    float fr = __expf((float)j * (-logf(10000.0f) / 63.0f));
    float e = ts[0] * fr;
    emb[j] = sinf(e);
    emb[j + 64] = cosf(e);
  }
  __syncthreads();
  float s = b[j];
  for (int i = 0; i < 128; ++i) s += emb[i] * W[i * 128 + j];
  tout[j] = fmaxf(s, 0.0f);
}

__global__ void te_kernel(const float* __restrict__ t, const float* __restrict__ W,
                          const float* __restrict__ b, float* __restrict__ te, int cout) {
  int j = blockIdx.x * blockDim.x + threadIdx.x;
  if (j >= cout) return;
  float s = b[j];
  for (int i = 0; i < 128; ++i) s += t[i] * W[i * cout + j];
  te[j] = silu_f(s);
}

// Weight f32[fi,fo] -> bf16 transposed [fo,fi]
__global__ void wconv_kernel(const float* __restrict__ W, bf16* __restrict__ Wt,
                             int fi, int fo) {
  int gid = blockIdx.x * blockDim.x + threadIdx.x;
  if (gid >= fi * fo) return;
  int o = gid / fi, i = gid - o * fi;
  Wt[gid] = (bf16)W[(size_t)i * fo + o];
}

// ---------------------------------------------------------------------------
// WMMA GEMM: out[N, cout] = A[N, cin](bf16) @ Wt[cout, cin]^T (+ bias) (epi)
// 64x64 block tile, 8 waves, each wave: one A frag x two B frags.
// Global->LDS via async copy (ASYNCcnt), double-buffered.
// ---------------------------------------------------------------------------
union FragBF { v16bf v; float4 f[2]; };

__global__ __launch_bounds__(256)
void wmma_gemm_kernel(const bf16* __restrict__ A, const bf16* __restrict__ Wt,
                      const float* __restrict__ bias,
                      float* __restrict__ outF, bf16* __restrict__ outB,
                      int N, int cin, int cout, int epi) {
  __shared__ bf16 As[2][64][40];   // row stride 80B (16B aligned)
  __shared__ bf16 Bs[2][64][40];
  const int tid  = threadIdx.x;
  const int lane = tid & 31;
  const int wave = tid >> 5;
  const int wm   = wave & 3;          // M tile 0..3
  const int wn   = (wave >> 2) << 1;  // N tiles {0,1} or {2,3}
  const int bm   = blockIdx.x * 64;
  const int bn   = blockIdx.y * 64;

  const int lrow = tid >> 2;          // 0..63
  const int lchk = (tid & 3) << 3;    // 0,8,16,24 (bf16 units)

  int arow = bm + lrow; if (arow >= N) arow = N - 1;
  const bf16* aptr = A  + (size_t)arow * cin + lchk;
  const bf16* bptr = Wt + (size_t)(bn + lrow) * cin + lchk;

  v8f acc0 = {}; v8f acc1 = {};
  const int fr  = lane & 15;
  const int koA = (lane >> 4) << 3;   // unused; A also reads K contiguous 16 at koB
  const int koB = (lane >> 4) << 4;   // B: K contiguous 16 at koB

  const int nk = cin >> 5;
  // prologue: stage tile 0 into buffer 0
  async_copy16(aptr, &As[0][lrow][lchk]);
  async_copy16(bptr, &Bs[0][lrow][lchk]);

  for (int kk = 0; kk < nk; ++kk) {
    const int buf = kk & 1;
    if (kk + 1 < nk) {
      const int k1 = (kk + 1) << 5;
      async_copy16(aptr + k1, &As[buf ^ 1][lrow][lchk]);
      async_copy16(bptr + k1, &Bs[buf ^ 1][lrow][lchk]);
      wait_async<2>();   // oldest pair (current buf) resident
    } else {
      wait_async<0>();
    }
    __syncthreads();
    FragBF fa, fb0, fb1;
    fa.f[0]  = *(const float4*)(&As[buf][wm * 16 + fr][koB]);
    fa.f[1]  = *(const float4*)(&As[buf][wm * 16 + fr][koB + 8]);
    fb0.f[0] = *(const float4*)(&Bs[buf][wn * 16 + fr][koB]);
    fb0.f[1] = *(const float4*)(&Bs[buf][wn * 16 + fr][koB + 8]);
    fb1.f[0] = *(const float4*)(&Bs[buf][(wn + 1) * 16 + fr][koB]);
    fb1.f[1] = *(const float4*)(&Bs[buf][(wn + 1) * 16 + fr][koB + 8]);
    acc0 = __builtin_amdgcn_wmma_f32_16x16x32_bf16(false, fa.v, false, fb0.v,
                                                   (short)0, acc0, false, false);
    acc1 = __builtin_amdgcn_wmma_f32_16x16x32_bf16(false, fa.v, false, fb1.v,
                                                   (short)0, acc1, false, false);
    __syncthreads();   // consumed buf safe to overwrite next iteration
  }

  const int n0 = bn + wn * 16 + fr;
  const int n1 = n0 + 16;
  const int mb = bm + wm * 16 + ((lane >> 4) << 3);
  const float b0v = bias ? bias[n0] : 0.0f;
  const float b1v = bias ? bias[n1] : 0.0f;
#pragma unroll
  for (int v = 0; v < 8; ++v) {
    int r = mb + v;
    if (r >= N) continue;
    float x0 = acc0[v] + b0v;
    float x1 = acc1[v] + b1v;
    if (epi == 1) { x0 = silu_f(x0); x1 = silu_f(x1); }
    size_t o0 = (size_t)r * cout + n0;
    size_t o1 = (size_t)r * cout + n1;
    if (outF) { outF[o0] = x0;       outF[o1] = x1; }
    if (outB) { outB[o0] = (bf16)x0; outB[o1] = (bf16)x1; }
  }
}

// ---------------------------------------------------------------------------
// Edge scatter: acc[dst] += g[src] * norm   (8 channels / thread)
// ---------------------------------------------------------------------------
__global__ void scatter_edges_kernel(const int* __restrict__ src, const int* __restrict__ dst,
                                     const float* __restrict__ nrm, const bf16* __restrict__ g,
                                     float* __restrict__ acc, int cshift) {
  const int cksh = cshift - 3;               // chunks = cout/8
  long gid = (long)blockIdx.x * blockDim.x + threadIdx.x;
  if (gid >= ((long)NE << cksh)) return;
  int e = (int)(gid >> cksh);
  int c = ((int)gid & ((1 << cksh) - 1)) << 3;
  int s = src[e], d = dst[e];
  float nm = nrm[e];
  const bf16* gp = g + ((size_t)s << cshift) + c;
  float* ap = acc + ((size_t)d << cshift) + c;
  float4 raw = *(const float4*)gp;           // 8 bf16
  const bf16* gv = (const bf16*)&raw;
#pragma unroll
  for (int j = 0; j < 8; ++j) atomicAdd(ap + j, (float)gv[j] * nm);
}

// out = silu(acc + g*dis^2 + bias) + te   (bf16)
__global__ void gcn_finalize_kernel(const float* __restrict__ acc, const bf16* __restrict__ g,
                                    const float* __restrict__ dis, const float* __restrict__ bias,
                                    const float* __restrict__ te, bf16* __restrict__ out,
                                    int cshift) {
  long gid = (long)blockIdx.x * blockDim.x + threadIdx.x;
  if (gid >= ((long)NN << cshift)) return;
  int row = (int)(gid >> cshift);
  int c = (int)gid & ((1 << cshift) - 1);
  float d = dis[row];
  float h = acc[gid] + (float)g[gid] * d * d + bias[c];
  out[gid] = (bf16)(silu_f(h) + te[c]);
}

// LayerNorm (one wave32 per row), input f32, output bf16
__global__ void ln_kernel(const float* __restrict__ zs, const float* __restrict__ gm,
                          const float* __restrict__ bt, bf16* __restrict__ out, int cout) {
  int wave = threadIdx.x >> 5, lane = threadIdx.x & 31;
  int row = blockIdx.x * 8 + wave;
  if (row >= NN) return;
  const float* zp = zs + (size_t)row * cout;
  float sum = 0.0f, sq = 0.0f;
  for (int c = lane; c < cout; c += 32) { float v = zp[c]; sum += v; sq += v * v; }
  for (int m = 16; m > 0; m >>= 1) {
    sum += __shfl_xor(sum, m, 32);
    sq  += __shfl_xor(sq,  m, 32);
  }
  float mean = sum / (float)cout;
  float var  = sq / (float)cout - mean * mean;
  float inv  = rsqrtf(var + 1e-5f);
  for (int c = lane; c < cout; c += 32)
    out[(size_t)row * cout + c] = (bf16)((zp[c] - mean) * inv * gm[c] + bt[c]);
}

// ---------------------------------------------------------------------------
extern "C" void kernel_launch(void* const* d_in, const int* in_sizes, int n_in,
                              void* d_out, int out_size, void* d_ws, size_t ws_size,
                              hipStream_t stream) {
  const float* x  = (const float*)d_in[0];
  const float* ts = (const float*)d_in[1];
  const int* ei   = (const int*)d_in[2];
  const int* src  = ei;
  const int* dst  = ei + NE;

  // jax pytree flatten order (dict keys sorted): down, latent, layer0, layer_out, time, up
  const float* P[64];
  for (int i = 3; i < n_in && i < 64; ++i) P[i] = (const float*)d_in[i];
  const float *d0gW=P[3], *d0gb=P[4], *d0lW=P[5], *d0lb=P[6], *d0lnb=P[7], *d0lng=P[8], *d0tW=P[9], *d0tb=P[10];
  const float *d1gW=P[11], *d1gb=P[12], *d1lW=P[13], *d1lb=P[14], *d1lnb=P[15], *d1lng=P[16], *d1tW=P[17], *d1tb=P[18];
  const float *lbW=P[19], *lbb=P[20], *loW=P[21], *lob_=P[22];
  const float *l0W=P[23], *l0b=P[24];
  const float *outW=P[25], *outb=P[26];
  const float *tmW=P[27], *tmb=P[28];
  const float *u0gW=P[29], *u0gb=P[30], *u0lW=P[31], *u0lb=P[32], *u0lnb=P[33], *u0lng=P[34], *u0tW=P[35], *u0tb=P[36];
  const float *u1gW=P[37], *u1gb=P[38], *u1lW=P[39], *u1lb=P[40], *u1lnb=P[41], *u1lng=P[42], *u1tW=P[43], *u1tb=P[44];

  size_t off = 0;
  auto take = [&](size_t bytes) -> char* {
    char* p = (char*)d_ws + off;
    off += (bytes + 255) & ~(size_t)255;
    return p;
  };
  float* acc  = (float*)take((size_t)NN * 256 * 4);  // f32 accumulator / pre-LN
  bf16*  gB   = (bf16*) take((size_t)NN * 256 * 2);  // GCN transform output
  bf16*  actA = (bf16*) take((size_t)NN * 256 * 2);
  bf16*  actB = (bf16*) take((size_t)NN * 256 * 2);
  float* deg  = (float*)take((size_t)NN * 4);
  float* dis  = (float*)take((size_t)NN * 4);
  float* nrm  = (float*)take((size_t)NE * 4);
  float* xmin = (float*)take(64 * 4);
  float* xmax = (float*)take(64 * 4);
  float* tvec = (float*)take(128 * 4);
  float* te0  = (float*)take(128 * 4);
  float* te1  = (float*)take(64 * 4);
  float* te2  = (float*)take(128 * 4);
  float* te3  = (float*)take(256 * 4);
  bf16* w0T  = (bf16*)take(256UL * 64 * 2);
  bf16* d0gT = (bf16*)take(128UL * 256 * 2);
  bf16* d0lT = (bf16*)take(128UL * 128 * 2);
  bf16* d1gT = (bf16*)take(64UL * 128 * 2);
  bf16* d1lT = (bf16*)take(64UL * 64 * 2);
  bf16* lbT  = (bf16*)take(128UL * 64 * 2);
  bf16* loT  = (bf16*)take(64UL * 128 * 2);
  bf16* u0gT = (bf16*)take(128UL * 64 * 2);
  bf16* u0lT = (bf16*)take(128UL * 128 * 2);
  bf16* u1gT = (bf16*)take(256UL * 128 * 2);
  bf16* u1lT = (bf16*)take(256UL * 256 * 2);
  bf16* woT  = (bf16*)take(64UL * 256 * 2);

  auto gemm = [&](const bf16* A, const bf16* Wt, const float* bias,
                  float* oF, bf16* oB, int cin, int cout, int epi) {
    dim3 g((NN + 63) / 64, cout / 64);
    wmma_gemm_kernel<<<g, 256, 0, stream>>>(A, Wt, bias, oF, oB, NN, cin, cout, epi);
  };
  auto wconv = [&](const float* W, bf16* Wt, int fi, int fo) {
    int n = fi * fo;
    wconv_kernel<<<(n + 255) / 256, 256, 0, stream>>>(W, Wt, fi, fo);
  };

  // ---- preprocessing ----
  colminmax_kernel<<<64, 256, 0, stream>>>(x, xmin, xmax);
  scale_x_kernel<<<((long)NN * NF + 255) / 256, 256, 0, stream>>>(x, xmin, xmax, actA);
  zero_f32_kernel<<<(NN + 255) / 256, 256, 0, stream>>>(deg, NN);
  deg_kernel<<<(NE + 255) / 256, 256, 0, stream>>>(dst, deg);
  dis_kernel<<<(NN + 255) / 256, 256, 0, stream>>>(deg, dis);
  norm_kernel<<<(NE + 255) / 256, 256, 0, stream>>>(src, dst, dis, nrm);
  time_kernel<<<1, 128, 0, stream>>>(ts, tmW, tmb, tvec);
  te_kernel<<<1, 128, 0, stream>>>(tvec, d0tW, d0tb, te0, 128);
  te_kernel<<<1, 64,  0, stream>>>(tvec, d1tW, d1tb, te1, 64);
  te_kernel<<<1, 128, 0, stream>>>(tvec, u0tW, u0tb, te2, 128);
  te_kernel<<<1, 256, 0, stream>>>(tvec, u1tW, u1tb, te3, 256);
  wconv(l0W,  w0T,  64, 256);
  wconv(d0gW, d0gT, 256, 128);
  wconv(d0lW, d0lT, 128, 128);
  wconv(d1gW, d1gT, 128, 64);
  wconv(d1lW, d1lT, 64, 64);
  wconv(lbW,  lbT,  64, 128);
  wconv(loW,  loT,  128, 64);
  wconv(u0gW, u0gT, 64, 128);
  wconv(u0lW, u0lT, 128, 128);
  wconv(u1gW, u1gT, 128, 256);
  wconv(u1lW, u1lT, 256, 256);
  wconv(outW, woT,  256, 64);

  // ---- layer0: h = scaled_x @ W0 + b0 ----
  gemm(actA, w0T, l0b, nullptr, actB, 64, 256, 0);
  bf16* cur = actB;   // live activation (bf16)
  bf16* alt = actA;   // scratch for h1p

  auto gblock = [&](const bf16* wgT, const float* gb, const bf16* wlT, const float* lb,
                    const float* lng, const float* lnb, const float* te,
                    int cin, int cout, int cshift) {
    gemm(cur, wgT, nullptr, nullptr, gB, cin, cout, 0);          // g = h @ Wg
    long nc = (long)NN << cshift;
    zero_f32_kernel<<<(unsigned)((nc + 255) / 256), 256, 0, stream>>>(acc, nc);
    long tot = (long)NE << (cshift - 3);
    scatter_edges_kernel<<<(unsigned)((tot + 255) / 256), 256, 0, stream>>>(src, dst, nrm, gB, acc, cshift);
    gcn_finalize_kernel<<<(unsigned)((nc + 255) / 256), 256, 0, stream>>>(acc, gB, dis, gb, te, alt, cshift);
    gemm(alt, wlT, lb, acc, nullptr, cout, cout, 1);             // silu(h1p @ Wl + bl) -> f32
    ln_kernel<<<(NN + 7) / 8, 256, 0, stream>>>(acc, lng, lnb, cur, cout);
  };

  gblock(d0gT, d0gb, d0lT, d0lb, d0lng, d0lnb, te0, 256, 128, 7);
  gblock(d1gT, d1gb, d1lT, d1lb, d1lng, d1lnb, te1, 128, 64, 6);

  // ---- latent ----
  gemm(cur, lbT, lbb,  nullptr, alt, 64, 128, 1);   // silu
  gemm(alt, loT, lob_, nullptr, cur, 128, 64, 0);

  gblock(u0gT, u0gb, u0lT, u0lb, u0lng, u0lnb, te2, 64, 128, 7);
  gblock(u1gT, u1gb, u1lT, u1lb, u1lng, u1lnb, te3, 128, 256, 8);

  // ---- output projection -> d_out f32 [NN, 64] ----
  gemm(cur, woT, outb, (float*)d_out, nullptr, 256, 64, 0);
}
